// GATEncoder_53618371723610
// MI455X (gfx1250) — matrix operand order repaired and checked by reading, not compiled, not run on hardware
//
#include <hip/hip_runtime.h>
#include <math.h>

typedef __attribute__((ext_vector_type(16))) __bf16 v16bf;
typedef __attribute__((ext_vector_type(8)))  __bf16 v8bf;
typedef __attribute__((ext_vector_type(4)))  __bf16 v4bf;
typedef __attribute__((ext_vector_type(8)))  float  v8f;

constexpr int GB  = 8;           // graphs (batch)
constexpr int GN  = 4096;        // nodes
constexpr int GD  = 256;         // embedding / per-head dim
constexpr int GH  = 8;           // heads
constexpr int GHD = GH * GD;     // 2048
constexpr int GE  = 131072;      // edges
constexpr int GEN = GE + GN;     // edges + self loops = 135168
constexpr float NEG_SLOPE = 0.2f;
constexpr float LN_EPS    = 1e-5f;

// ---------------------------------------------------------------------------
// Generic GEMM via bf16 WMMA:  C[M,Nc] = A[M,K] * B[Nc,K]^T
// A,C strided per graph (blockIdx.z); B shared across graphs.
// ABIAS: add aBias[k] to A during staging; CBIAS: add cBias[col] at store.
// Block = 256 threads = 8 waves; tile 128x128; each wave owns 32(M)x64(N).
// Double-buffered LDS: next tile's global loads issue before current WMMAs.
// ---------------------------------------------------------------------------
constexpr int TM = 128, TN = 128, TK = 32;
constexpr int LSTR = 40;   // LDS row stride in bf16 (80 B -> keeps 16B alignment)

template<bool ABIAS>
__device__ __forceinline__ void load_tile_regs(
    const float* __restrict__ Ag, const float* __restrict__ Bm,
    int m0, int n0, int K, int k0, int lrow, int lcol,
    const float* __restrict__ aBias, float4 (&ra)[4], float4 (&rb)[4])
{
  #pragma unroll
  for (int p = 0; p < 4; ++p) {
    const int r = lrow + p * 32;
    ra[p] = *(const float4*)(Ag + (long)(m0 + r) * K + k0 + lcol);
    rb[p] = *(const float4*)(Bm + (long)(n0 + r) * K + k0 + lcol);
  }
  if (ABIAS) {
    const float b0 = aBias[k0 + lcol + 0], b1 = aBias[k0 + lcol + 1];
    const float b2 = aBias[k0 + lcol + 2], b3 = aBias[k0 + lcol + 3];
    #pragma unroll
    for (int p = 0; p < 4; ++p) {
      ra[p].x += b0; ra[p].y += b1; ra[p].z += b2; ra[p].w += b3;
    }
  }
}

__device__ __forceinline__ void store_tile_lds(
    __bf16* As, __bf16* Bs, int lrow, int lcol,
    const float4 (&ra)[4], const float4 (&rb)[4])
{
  #pragma unroll
  for (int p = 0; p < 4; ++p) {
    const int r = lrow + p * 32;
    v4bf pa; pa[0]=(__bf16)ra[p].x; pa[1]=(__bf16)ra[p].y;
             pa[2]=(__bf16)ra[p].z; pa[3]=(__bf16)ra[p].w;
    *(v4bf*)(&As[r * LSTR + lcol]) = pa;
    v4bf pb; pb[0]=(__bf16)rb[p].x; pb[1]=(__bf16)rb[p].y;
             pb[2]=(__bf16)rb[p].z; pb[3]=(__bf16)rb[p].w;
    *(v4bf*)(&Bs[r * LSTR + lcol]) = pb;
  }
}

template<bool ABIAS, bool CBIAS>
__global__ __launch_bounds__(256) void gemm_bf16_wmma(
    const float* __restrict__ A, const float* __restrict__ Bm, float* __restrict__ C,
    int Nc, int K, long aStride, long cStride,
    const float* __restrict__ aBias, const float* __restrict__ cBias)
{
  __shared__ __bf16 As[2][TM * LSTR];
  __shared__ __bf16 Bs[2][TN * LSTR];

  const float* Ag = A + (long)blockIdx.z * aStride;
  float*       Cg = C + (long)blockIdx.z * cStride;
  const int m0   = blockIdx.x * TM;
  const int n0   = blockIdx.y * TN;
  const int t    = threadIdx.x;
  const int lane = t & 31, wave = t >> 5;
  const int wm   = wave & 3;          // 4 waves cover 128 rows (32 each)
  const int wn   = wave >> 2;         // 2 waves cover 128 cols (64 each)
  const int hl   = lane >> 4;         // which 16-lane half
  const int lr   = lane & 15;
  const int lrow = t >> 3;            // 0..31
  const int lcol = (t & 7) * 4;       // 0..28 step 4

  v8f acc[2][4];
  #pragma unroll
  for (int i = 0; i < 2; ++i)
    #pragma unroll
    for (int j = 0; j < 4; ++j) {
      v8f z = {0.f,0.f,0.f,0.f,0.f,0.f,0.f,0.f};
      acc[i][j] = z;
    }

  // prologue: stage tile 0
  float4 ra[4], rb[4];
  load_tile_regs<ABIAS>(Ag, Bm, m0, n0, K, 0, lrow, lcol, aBias, ra, rb);
  store_tile_lds(As[0], Bs[0], lrow, lcol, ra, rb);

  int cur = 0;
  for (int k0 = 0; k0 < K; k0 += TK) {
    __syncthreads();   // buf[cur] stores visible; buf[cur^1] readers done
    const bool hasNext = (k0 + TK) < K;
    if (hasNext)       // issue next tile's global loads early (latency hiding)
      load_tile_regs<ABIAS>(Ag, Bm, m0, n0, K, k0 + TK, lrow, lcol, aBias, ra, rb);

    // A fragment (16x32 bf16): lane half hl, row lr.
    // elems 0..7 = K(hl*8 .. +7), elems 8..15 = K(16+hl*8 .. +7)
    v16bf af[2];
    #pragma unroll
    for (int i = 0; i < 2; ++i) {
      const __bf16* base = &As[cur][(wm * 32 + i * 16 + lr) * LSTR];
      ((v8bf*)&af[i])[0] = *(const v8bf*)(base + hl * 8);
      ((v8bf*)&af[i])[1] = *(const v8bf*)(base + 16 + hl * 8);
    }
    // B fragment (32x16 bf16): lane col lr, elems 0..7 = K(hl*8 .. +7), elems 8..15 = K(16+hl*8 .. +7)
    v16bf bfr[4];
    #pragma unroll
    for (int j = 0; j < 4; ++j) {
      const __bf16* base = &Bs[cur][(wn * 64 + j * 16 + lr) * LSTR];
      ((v8bf*)&bfr[j])[0] = *(const v8bf*)(base + hl * 8);
      ((v8bf*)&bfr[j])[1] = *(const v8bf*)(base + 16 + hl * 8);
    }

    #pragma unroll
    for (int i = 0; i < 2; ++i)
      #pragma unroll
      for (int j = 0; j < 4; ++j)
        acc[i][j] = __builtin_amdgcn_wmma_f32_16x16x32_bf16(
            false, af[i], false, bfr[j], (short)0, acc[i][j], false, false);

    if (hasNext)
      store_tile_lds(As[cur ^ 1], Bs[cur ^ 1], lrow, lcol, ra, rb);
    cur ^= 1;
  }

  // epilogue: C-layout VGPR v -> row = base + hl*8 + v, col = base + lr
  #pragma unroll
  for (int i = 0; i < 2; ++i) {
    const int rbase = m0 + wm * 32 + i * 16 + hl * 8;
    #pragma unroll
    for (int j = 0; j < 4; ++j) {
      const int col = n0 + wn * 64 + j * 16 + lr;
      const float cb = CBIAS ? cBias[col] : 0.0f;
      #pragma unroll
      for (int v = 0; v < 8; ++v)
        Cg[(long)(rbase + v) * Nc + col] = acc[i][j][v] + cb;
    }
  }
}

// ---------------------------------------------------------------------------
// a_s[n,h] = sum_d h[n,h,d]*att_src[h,d] ;  a_d likewise.  Wave per node.
// ---------------------------------------------------------------------------
__global__ __launch_bounds__(256) void attn_coef(
    const float* __restrict__ hmat, const float* __restrict__ asrc,
    const float* __restrict__ adst, float* __restrict__ oas, float* __restrict__ oad)
{
  const int wave = threadIdx.x >> 5, lane = threadIdx.x & 31;
  const long node = (long)blockIdx.x * 8 + wave;      // flat b*GN+n
  const float* hrow = hmat + node * GHD;
  #pragma unroll
  for (int hh = 0; hh < GH; ++hh) {
    const float* hp = hrow + hh * GD;
    const float* ap = asrc + hh * GD;
    const float* bp = adst + hh * GD;
    float ss = 0.f, sd = 0.f;
    #pragma unroll
    for (int i = 0; i < GD / 32; ++i) {
      const float hv = hp[lane + 32 * i];
      ss += hv * ap[lane + 32 * i];
      sd += hv * bp[lane + 32 * i];
    }
    #pragma unroll
    for (int o = 16; o > 0; o >>= 1) {
      ss += __shfl_down(ss, o, 32);
      sd += __shfl_down(sd, o, 32);
    }
    if (lane == 0) { oas[node * GH + hh] = ss; oad[node * GH + hh] = sd; }
  }
}

// ---------------------------------------------------------------------------
// edge softmax helpers
// ---------------------------------------------------------------------------
__device__ __forceinline__ void atomicMaxF(float* addr, float val) {
  // exact for -inf-initialized buffers; avoids CAS loops
  if (val >= 0.f) atomicMax((int*)addr, __float_as_int(val));
  else            atomicMin((unsigned int*)addr, __float_as_uint(val));
}

__device__ __forceinline__ void edge_endpoints(const int* __restrict__ edges,
                                               long idx, int& b, int& src, int& dst) {
  b = (int)(idx / GEN);
  const int e = (int)(idx % GEN);
  const int* eb = edges + (long)b * 2 * GE;
  src = (e < GE) ? eb[e]      : (e - GE);   // self loops appended
  dst = (e < GE) ? eb[GE + e] : (e - GE);
}

__global__ __launch_bounds__(256) void edge_max_k(
    const int* __restrict__ edges, const float* __restrict__ as_,
    const float* __restrict__ ad_, float* __restrict__ emax)
{
  const long idx = (long)blockIdx.x * 256 + threadIdx.x;
  int b, src, dst; edge_endpoints(edges, idx, b, src, dst);
  const float* sp = as_ + ((long)b * GN + src) * GH;
  const float* dp = ad_ + ((long)b * GN + dst) * GH;
  float* em = emax + ((long)b * GN + dst) * GH;
  #pragma unroll
  for (int hh = 0; hh < GH; ++hh) {
    float e = sp[hh] + dp[hh];
    e = (e > 0.f) ? e : NEG_SLOPE * e;
    atomicMaxF(&em[hh], e);
  }
}

__global__ __launch_bounds__(256) void edge_exp_k(
    const int* __restrict__ edges, const float* __restrict__ as_,
    const float* __restrict__ ad_, const float* __restrict__ emax,
    float* __restrict__ denom, float* __restrict__ alphab)
{
  const long idx = (long)blockIdx.x * 256 + threadIdx.x;
  int b, src, dst; edge_endpoints(edges, idx, b, src, dst);
  const float* sp = as_ + ((long)b * GN + src) * GH;
  const float* dp = ad_ + ((long)b * GN + dst) * GH;
  const float* em = emax + ((long)b * GN + dst) * GH;
  float* dn = denom + ((long)b * GN + dst) * GH;
  #pragma unroll
  for (int hh = 0; hh < GH; ++hh) {
    float e = sp[hh] + dp[hh];
    e = (e > 0.f) ? e : NEG_SLOPE * e;
    const float ex = __expf(e - em[hh]);
    alphab[idx * GH + hh] = ex;
    atomicAdd(&dn[hh], ex);
  }
}

// one block per edge; wave hh handles head hh; lanes stream 256 channels
__global__ __launch_bounds__(256) void aggregate_k(
    const int* __restrict__ edges, const float* __restrict__ hmat,
    const float* __restrict__ alphab, const float* __restrict__ denom,
    float* __restrict__ outg)
{
  const long idx = blockIdx.x;                 // b*GEN + e
  int b, src, dst; edge_endpoints(edges, idx, b, src, dst);
  const int hh = threadIdx.x >> 5, lane = threadIdx.x & 31;
  const float alpha = alphab[idx * GH + hh] /
                      (denom[((long)b * GN + dst) * GH + hh] + 1e-16f);
  const float* hs = hmat + ((long)b * GN + src) * GHD + hh * GD;
  float*       od = outg + ((long)b * GN + dst) * GHD + hh * GD;
  #pragma unroll
  for (int i = 0; i < GD / 32; ++i)
    atomicAdd(&od[lane + 32 * i], hs[lane + 32 * i] * alpha);
}

// ---------------------------------------------------------------------------
// fused LayerNorm + residual + ReLU, wave per node
// ---------------------------------------------------------------------------
__global__ __launch_bounds__(256) void ln_res_relu(
    const float* __restrict__ g2, const float* __restrict__ res,
    const float* __restrict__ gamma, const float* __restrict__ beta,
    float* __restrict__ outp)
{
  const int wave = threadIdx.x >> 5, lane = threadIdx.x & 31;
  const long node = (long)blockIdx.x * 8 + wave;
  const float* gp = g2 + node * GD;
  float x[GD / 32];
  float s = 0.f;
  #pragma unroll
  for (int i = 0; i < GD / 32; ++i) { x[i] = gp[lane + 32 * i]; s += x[i]; }
  #pragma unroll
  for (int o = 16; o > 0; o >>= 1) s += __shfl_xor(s, o, 32);
  const float mu = s * (1.0f / GD);
  float v = 0.f;
  #pragma unroll
  for (int i = 0; i < GD / 32; ++i) { const float dv = x[i] - mu; v += dv * dv; }
  #pragma unroll
  for (int o = 16; o > 0; o >>= 1) v += __shfl_xor(v, o, 32);
  const float rs = rsqrtf(v * (1.0f / GD) + LN_EPS);
  #pragma unroll
  for (int i = 0; i < GD / 32; ++i) {
    const int d = lane + 32 * i;
    const float y = (x[i] - mu) * rs * gamma[d] + beta[d] + res[node * GD + d];
    outp[node * GD + d] = fmaxf(y, 0.f);
  }
}

// ---------------------------------------------------------------------------
// init kernels (scratch must be re-initialized every call: atomics accumulate)
// ---------------------------------------------------------------------------
__global__ __launch_bounds__(256) void zero_f4(float4* p) {
  const size_t i = (size_t)blockIdx.x * 256 + threadIdx.x;
  p[i] = make_float4(0.f, 0.f, 0.f, 0.f);
}
__global__ __launch_bounds__(256) void init_softmax(float* emax, float* denom) {
  const size_t i = (size_t)blockIdx.x * 256 + threadIdx.x;
  emax[i]  = -INFINITY;
  denom[i] = 0.f;
}

// ---------------------------------------------------------------------------
extern "C" void kernel_launch(void* const* d_in, const int* in_sizes, int n_in,
                              void* d_out, int out_size, void* d_ws, size_t ws_size,
                              hipStream_t stream)
{
  (void)in_sizes; (void)n_in; (void)out_size; (void)ws_size;
  const float* x0    = (const float*)d_in[0];   // [B,N,D]
  const int*   edges = (const int*)  d_in[1];   // [B,2,E] int32
  const float* Wl    = (const float*)d_in[2];   // [L,HD,D]
  const float* asrc  = (const float*)d_in[3];   // [L,H,D]
  const float* adst  = (const float*)d_in[4];   // [L,H,D]
  const float* gbias = (const float*)d_in[5];   // [L,HD]
  const float* linW  = (const float*)d_in[6];   // [L,D,HD]
  const float* linb  = (const float*)d_in[7];   // [L,D]
  const float* gamma = (const float*)d_in[8];   // [D]
  const float* beta  = (const float*)d_in[9];   // [D]
  float* out = (float*)d_out;                   // [B,N,D]

  char* ws = (char*)d_ws;
  size_t off = 0;
  auto take = [&](size_t bytes) -> char* {
    char* p = ws + off; off += (bytes + 255) & ~(size_t)255; return p;
  };
  float* hbuf   = (float*)take((size_t)GB * GN * GHD * 4);  // 256 MB
  float* gatout = (float*)take((size_t)GB * GN * GHD * 4);  // 256 MB
  float* alphab = (float*)take((size_t)GB * GEN * GH * 4);  //  33 MB
  float* a_s    = (float*)take((size_t)GB * GN * GH * 4);
  float* a_d    = (float*)take((size_t)GB * GN * GH * 4);
  float* emax   = (float*)take((size_t)GB * GN * GH * 4);
  float* denom  = (float*)take((size_t)GB * GN * GH * 4);
  float* g2     = (float*)take((size_t)GB * GN * GD * 4);   //  32 MB
  float* cur    = (float*)take((size_t)GB * GN * GD * 4);   //  32 MB

  for (int l = 0; l < 2; ++l) {
    const float* xin  = (l == 0) ? x0  : cur;
    float*       xout = (l == 1) ? out : cur;

    zero_f4<<<dim3(((size_t)GB * GN * GHD / 4) / 256), 256, 0, stream>>>((float4*)gatout);
    init_softmax<<<dim3((GB * GN * GH) / 256), 256, 0, stream>>>(emax, denom);

    // h = x @ W[l]^T     [N,256] x [2048,256]^T -> [N,2048]
    gemm_bf16_wmma<false, false><<<dim3(GN / TM, GHD / TN, GB), 256, 0, stream>>>(
        xin, Wl + (size_t)l * GHD * GD, hbuf, GHD, GD,
        (long)GN * GD, (long)GN * GHD, nullptr, nullptr);

    attn_coef<<<dim3(GB * GN / 8), 256, 0, stream>>>(
        hbuf, asrc + (size_t)l * GH * GD, adst + (size_t)l * GH * GD, a_s, a_d);

    edge_max_k<<<dim3(GB * GEN / 256), 256, 0, stream>>>(edges, a_s, a_d, emax);
    edge_exp_k<<<dim3(GB * GEN / 256), 256, 0, stream>>>(edges, a_s, a_d, emax, denom, alphab);
    aggregate_k<<<dim3(GB * GEN), 256, 0, stream>>>(edges, hbuf, alphab, denom, gatout);

    // g = (gat + gat_bias) @ lin_W[l]^T + lin_b   [N,2048] x [256,2048]^T -> [N,256]
    gemm_bf16_wmma<true, true><<<dim3(GN / TM, GD / TN, GB), 256, 0, stream>>>(
        gatout, linW + (size_t)l * GD * GHD, g2, GD, GHD,
        (long)GN * GHD, (long)GN * GD,
        gbias + (size_t)l * GHD, linb + (size_t)l * GD);

    ln_res_relu<<<dim3(GB * GN / 8), 256, 0, stream>>>(g2, xin, gamma, beta, xout);
  }
}
